// VanillaRNN_78365973283507
// MI455X (gfx1250) — matrix operand order repaired and checked by reading, not compiled, not run on hardware
//
#include <hip/hip_runtime.h>

#define B_   512
#define T_   512
#define D_   128
#define H_   256
#define C_   10
#define KTOT 384          // D_ + H_ : concatenated [x_t | h_{t-1}] K dimension
#define KP   392          // padded row pitch (bf16 elems): 196 dwords -> conflict-free
#define BT   16           // batch rows per workgroup (WMMA M)
#define NTHREADS 512      // 16 waves; one 16-wide N tile per wave
#define ABUF (BT * KP)    // one activation panel (bf16 elems)

typedef __bf16 bf16_t;
typedef __attribute__((ext_vector_type(4)))  __bf16 v4bf;
typedef __attribute__((ext_vector_type(8)))  __bf16 v8bf;
typedef __attribute__((ext_vector_type(16))) __bf16 v16bf;
typedef __attribute__((ext_vector_type(8)))  float  v8f;

static __device__ __forceinline__ v16bf concat8(v8bf lo, v8bf hi) {
  return __builtin_shufflevector(lo, hi, 0,1,2,3,4,5,6,7,8,9,10,11,12,13,14,15);
}

static __device__ __forceinline__ float fast_tanh(float v) {
#if __has_builtin(__builtin_amdgcn_tanhf)
  return __builtin_amdgcn_tanhf(v);          // V_TANH_F32 (TRANS)
#else
  float c = fminf(fmaxf(v, -15.0f), 15.0f);  // branch-free fallback
  float e = __expf(2.0f * c);                // v_exp_f32
  return (e - 1.0f) * __builtin_amdgcn_rcpf(e + 1.0f);
#endif
}

__global__ __launch_bounds__(NTHREADS, 1) void rnn_fused_kernel(
    const float* __restrict__ x,   const float* __restrict__ Whx,
    const float* __restrict__ Whh, const float* __restrict__ Wph,
    const float* __restrict__ bh,  const float* __restrict__ bpv,
    float* __restrict__ out)
{
  extern __shared__ char smem_raw[];
  // sW : [H_][KP] bf16, transposed weights: sW[n][k] = (k<D_? Whx[k][n] : Whh[k-D_][n])
  // sA : two ping-pong [BT][KP] activation panels: cols 0..127 = x_t, 128..383 = h_{t-1}
  bf16_t* sW  = (bf16_t*)smem_raw;
  bf16_t* sA  = (bf16_t*)(smem_raw + (size_t)H_ * KP * 2);
  float*  sBh = (float*) (smem_raw + (size_t)H_ * KP * 2 + (size_t)2 * ABUF * 2);

  const int tid  = threadIdx.x;
  const int lane = tid & 31;
  const int wid  = tid >> 5;       // 16 waves
  const int half = lane >> 4;      // lane half (WMMA layout)
  const int lr   = lane & 15;
  const int b0   = blockIdx.x * BT;

  // ---- one-time init: weights -> LDS (bf16, transposed), bias, h0 = 0 ----
  {
    const int n   = tid >> 1;      // 0..255 : output column of W
    const int sub = tid & 1;
    for (int k = sub * 64; k < sub * 64 + 64; ++k)
      sW[n * KP + k] = (bf16_t)Whx[k * H_ + n];
    for (int k = sub * 128; k < sub * 128 + 128; ++k)
      sW[n * KP + D_ + k] = (bf16_t)Whh[k * H_ + n];
    if (tid < H_) sBh[tid] = bh[tid];
    for (int idx = tid; idx < BT * H_; idx += NTHREADS) {
      int r = idx >> 8, c = idx & 255;
      sA[r * KP + D_ + c] = (bf16_t)0.0f;   // h_{-1} = 0 in panel 0
    }
  }

  const int   n0 = wid * 16;       // this wave's N tile

  // x staging: wave w owns row w; each lane converts/stores 4 elements.
  const int xr = wid;              // 0..15 row of the batch tile
  const int xc = lane * 4;         // 0..124
  const float* xbase = x + (size_t)(b0 + xr) * T_ * D_ + xc;

  // prologue: x_0 -> panel 0; preload x_1 into registers
  {
    float4 x0 = *(const float4*)xbase;
    v4bf xv;
    xv[0] = (bf16_t)x0.x; xv[1] = (bf16_t)x0.y;
    xv[2] = (bf16_t)x0.z; xv[3] = (bf16_t)x0.w;
    *(v4bf*)(sA + xr * KP + xc) = xv;
  }
  float4 xf = *(const float4*)(xbase + D_);     // x_1
  __syncthreads();
  const float bv = sBh[n0 + lr];   // bias, hoisted out of the recurrence

  // ---- weight (B) fragments: VGPR-resident for all 512 steps ----
  v16bf breg[12];
  #pragma unroll
  for (int kk = 0; kk < 12; ++kk) {
    const bf16_t* bp = sW + (n0 + lr) * KP + kk * 32 + half * 8;
    breg[kk] = concat8(*(const v8bf*)bp, *(const v8bf*)(bp + 16));
  }

#define LOAD_A(buf, kk)                                                        \
  concat8(*(const v8bf*)((buf) + lr * KP + (kk) * 32 + half * 8),              \
          *(const v8bf*)((buf) + lr * KP + (kk) * 32 + half * 8 + 16))

  for (int t = 0; t < T_; ++t) {
    bf16_t* bufP = sA + (t & 1) * ABUF;          // read  panel: [x_t | h_{t-1}]
    bf16_t* bufQ = sA + ((t + 1) & 1) * ABUF;    // write panel: [x_{t+1} | h_t]
    __syncthreads();   // panel P fully written (t-1), panel Q fully read (t-1)

    v16bf areg[12];
    #pragma unroll
    for (int kk = 0; kk < 12; ++kk)
      areg[kk] = LOAD_A(bufP, kk);

    v8f acc0 = {bv, bv, bv, bv, bv, bv, bv, bv};
    v8f acc1 = {};
    #pragma unroll
    for (int kk = 0; kk < 12; ++kk) {
      if (kk & 1)
        acc1 = __builtin_amdgcn_wmma_f32_16x16x32_bf16(false, areg[kk], false, breg[kk],
                                                       (short)0, acc1, false, false);
      else
        acc0 = __builtin_amdgcn_wmma_f32_16x16x32_bf16(false, areg[kk], false, breg[kk],
                                                       (short)0, acc0, false, false);
    }

#if __has_builtin(__builtin_amdgcn_sched_group_barrier)
    // Pin the pipeline: keep a constant 3-fragment (6 ds_read) lead over the
    // WMMA drain so every s_wait_dscnt covers loads issued ~3 WMMAs earlier.
    __builtin_amdgcn_sched_group_barrier(0x100, 6, 0);   // 6 ds_read
    #pragma unroll
    for (int i = 0; i < 9; ++i) {
      __builtin_amdgcn_sched_group_barrier(0x008, 1, 0); // 1 wmma
      __builtin_amdgcn_sched_group_barrier(0x100, 2, 0); // 2 ds_read
    }
    __builtin_amdgcn_sched_group_barrier(0x008, 3, 0);   // 3 wmma (tail)
#endif

    acc0 += acc1;

    // write phase -> panel Q: x_{t+1} (and issue global load of x_{t+2}), h_t
    if (t + 1 < T_) {
      v4bf xv;
      xv[0] = (bf16_t)xf.x; xv[1] = (bf16_t)xf.y;
      xv[2] = (bf16_t)xf.z; xv[3] = (bf16_t)xf.w;
      *(v4bf*)(bufQ + xr * KP + xc) = xv;
      if (t + 2 < T_)
        xf = *(const float4*)(xbase + (size_t)(t + 2) * D_);
    }
    #pragma unroll
    for (int v = 0; v < 8; ++v)
      bufQ[(v + 8 * half) * KP + D_ + n0 + lr] = (bf16_t)fast_tanh(acc0[v]);
  }
  __syncthreads();

  // readout: h_{T-1} lives in panel (T_ & 1) == 0
  if (tid < BT * C_) {
    const int r = tid / C_, c = tid % C_;
    float s = bpv[c];
    for (int k = 0; k < H_; ++k)
      s += (float)sA[(T_ & 1) * ABUF + r * KP + D_ + k] * Wph[k * C_ + c];
    out[(size_t)(b0 + r) * C_ + c] = s;
  }
#undef LOAD_A
}

extern "C" void kernel_launch(void* const* d_in, const int* in_sizes, int n_in,
                              void* d_out, int out_size, void* d_ws, size_t ws_size,
                              hipStream_t stream) {
  const float* x   = (const float*)d_in[0];
  const float* Whx = (const float*)d_in[1];
  const float* Whh = (const float*)d_in[2];
  const float* Wph = (const float*)d_in[3];
  const float* bh  = (const float*)d_in[4];
  const float* bpv = (const float*)d_in[5];
  float* out = (float*)d_out;

  const size_t smem = (size_t)H_ * KP * 2 + (size_t)2 * ABUF * 2 + H_ * 4; // ~222 KB
  hipFuncSetAttribute((const void*)rnn_fused_kernel,
                      hipFuncAttributeMaxDynamicSharedMemorySize, (int)smem);
  rnn_fused_kernel<<<B_ / BT, NTHREADS, smem, stream>>>(x, Whx, Whh, Wph, bh, bpv, out);
}
